// TinyLocalWindowAttention_15075335209664
// MI455X (gfx1250) — matrix operand order repaired and checked by reading, not compiled, not run on hardware
//
#include <hip/hip_runtime.h>
#include <hip/hip_bf16.h>

typedef __attribute__((ext_vector_type(16))) __bf16 v16bf;
typedef __attribute__((ext_vector_type(8)))  __bf16 v8bf;
typedef __attribute__((ext_vector_type(8)))  float  v8f;

#define DIM   256
#define NHEADS 4
#define KD    16
#define VD    32
#define WSZ   7
#define NTOK  49
#define HW    112
#define HWHW  12544
#define ATTN_SCALE 0.25f

// ---- WMMA fragment loaders (wave32, bf16 16x16x32) ----
// A-matrix 16x32 (MxK): lane<16 -> row M=lane, K = {k0..k0+7, k0+16..k0+23}
//                       lane>=16 -> row M=lane-16, K = {k0+8..k0+15, k0+24..k0+31}
__device__ __forceinline__ v16bf load_a_frag(const __bf16* base, int stride,
                                             int lane, int m0, int k0) {
  const int half = lane >> 4;
  const int mr   = lane & 15;
  const __bf16* p = base + (m0 + mr) * stride + k0 + half * 8;
  v16bf r;
  ((v8bf*)&r)[0] = *(const v8bf*)(p);
  ((v8bf*)&r)[1] = *(const v8bf*)(p + 16);
  return r;
}

// B-matrix 32x16 (KxN), source stored n-major [n][k] (K contiguous):
// lane<16 -> col N=lane, K = {k0..k0+7, k0+16..k0+23} ; lane>=16 -> K = {k0+8..k0+15, k0+24..k0+31}
__device__ __forceinline__ v16bf load_b_frag(const __bf16* base, int stride,
                                             int lane, int n0, int k0) {
  const int half = lane >> 4;
  const int nr   = lane & 15;
  const __bf16* p = base + (n0 + nr) * stride + k0 + half * 8;
  v16bf r;
  ((v8bf*)&r)[0] = *(const v8bf*)(p);
  ((v8bf*)&r)[1] = *(const v8bf*)(p + 16);
  return r;
}

__device__ __forceinline__ v8f wmma_bf16(v16bf a, v16bf b, v8f c) {
  return __builtin_amdgcn_wmma_f32_16x16x32_bf16(false, a, false, b,
                                                 (short)0, c, false, false);
}

// ---- weight prep: fp32 [in][out] -> bf16 [out][in] (K-contiguous for B frags) ----
__global__ void prep_weights(const float* __restrict__ Wqkv,
                             const float* __restrict__ Wproj,
                             __bf16* __restrict__ WqkvT,
                             __bf16* __restrict__ WprojT) {
  int e = blockIdx.x * 256 + threadIdx.x;
  if (e < 256 * 256) {
    int o = e >> 8, i = e & 255;
    WqkvT[o * 256 + i] = (__bf16)Wqkv[i * 256 + o];
  } else {
    e -= 256 * 256;
    if (e < 128 * 256) {
      int o = e >> 7, i = e & 127;            // Wproj is [in=128][out=256]
      WprojT[o * 128 + i] = (__bf16)Wproj[i * 256 + o];
    }
  }
}

__global__ void __launch_bounds__(256)
winattn_kernel(const float* __restrict__ x,
               const float* __restrict__ bqkv,
               const float* __restrict__ bproj,
               const __bf16* __restrict__ WqkvT,
               const __bf16* __restrict__ WprojT,
               float* __restrict__ out) {
  // token-major activations; strides padded (odd multiples of 16B) vs 64 banks
  __shared__ __align__(16) __bf16 Xs[64 * 264];   // X  [tok][c]        33.8 KB
  __shared__ __align__(16) __bf16 Qs[64 * 136];   // Q  [tok][h*32+d]   17.4 KB (d>=16 zero)
  __shared__ __align__(16) __bf16 Ks[64 * 136];   // K  [tok][h*32+d]   17.4 KB (d>=16 zero)
  __shared__ __align__(16) __bf16 VTs[128 * 72];  // V^T[h*32+d][tok]   18.4 KB
  __shared__ __align__(16) float  Ss[64 * 65];    // logits             16.6 KB
  __shared__ __align__(16) __bf16 Ps[64 * 72];    // probs bf16          9.2 KB
  __shared__ __align__(16) __bf16 Os[64 * 136];   // attn out [tok][c]  17.4 KB

  const int tid  = threadIdx.x;
  const int wid  = tid >> 5;
  const int lane = tid & 31;
  const int half = lane >> 4;
  const int lr   = lane & 15;

  const int win = blockIdx.x;           // 8192 windows
  const int b   = win >> 8;
  const int wh  = (win >> 4) & 15;
  const int ww  = win & 15;
  const int h0  = wh * WSZ, w0 = ww * WSZ;

  // zero the K-dim pad (d = 16..31 of each head) of Q and K once
  for (int e = tid; e < 64 * 64; e += 256) {
    int row = e >> 6, c = e & 63;
    int col = ((c >> 4) << 5) + 16 + (c & 15);
    Qs[row * 136 + col] = (__bf16)0.0f;
    Ks[row * 136 + col] = (__bf16)0.0f;
  }

  // gather 7x7 window, all 256 channels -> bf16 LDS; rows 49..63 zero
  const float* xb = x + (size_t)b * DIM * HWHW;
  for (int e = tid; e < 64 * DIM; e += 256) {
    int c = e >> 6, m = e & 63;
    __bf16 v = (__bf16)0.0f;
    if (m < NTOK) {
      int i = (m * 9363) >> 16;           // m / 7 for m < 64
      int j = m - i * 7;
      v = (__bf16)xb[(size_t)c * HWHW + (h0 + i) * HW + (w0 + j)];
    }
    Xs[m * 264 + c] = v;
  }
  __syncthreads();

  // ---------- QKV = X @ Wqkv + bqkv ----------
  {
    v8f acc[2][4] = {};
    const int nb = wid * 32;                     // this wave's 32 output cols
#pragma unroll
    for (int ks = 0; ks < 8; ++ks) {
      v16bf bf0 = load_b_frag(WqkvT, 256, lane, nb,      ks * 32);
      v16bf bf1 = load_b_frag(WqkvT, 256, lane, nb + 16, ks * 32);
#pragma unroll
      for (int mt = 0; mt < 4; ++mt) {
        v16bf af = load_a_frag(Xs, 264, lane, mt * 16, ks * 32);
        acc[0][mt] = wmma_bf16(af, bf0, acc[0][mt]);
        acc[1][mt] = wmma_bf16(af, bf1, acc[1][mt]);
      }
    }
    // route to Q (scaled) / K / V^T staging
#pragma unroll
    for (int nt = 0; nt < 2; ++nt) {
      const int n = nb + nt * 16 + lr;
      const float bias = bqkv[n];
#pragma unroll
      for (int mt = 0; mt < 4; ++mt) {
#pragma unroll
        for (int r = 0; r < 8; ++r) {
          const int m = mt * 16 + half * 8 + r;
          const float v = acc[nt][mt][r] + bias;
          if (n < 64) {
            Qs[m * 136 + ((n >> 4) << 5) + (n & 15)] = (__bf16)(v * ATTN_SCALE);
          } else if (n < 128) {
            const int n2 = n - 64;
            Ks[m * 136 + ((n2 >> 4) << 5) + (n2 & 15)] = (__bf16)v;
          } else {
            VTs[(n - 128) * 72 + m] = (__bf16)v;
          }
        }
      }
    }
  }
  __syncthreads();

  // ---------- per-head attention ----------
#pragma unroll 1
  for (int h = 0; h < NHEADS; ++h) {
    // S = Q_h @ K_h^T  (scale folded into Q); 16 tiles, 2 per wave
#pragma unroll
    for (int t = 0; t < 2; ++t) {
      const int tile = wid * 2 + t;
      const int mt = tile >> 2, nt = tile & 3;
      v16bf af = load_a_frag(Qs, 136, lane, mt * 16, h * 32);
      v16bf bf = load_b_frag(Ks, 136, lane, nt * 16, h * 32);
      v8f c = {};
      c = wmma_bf16(af, bf, c);
      const int n = nt * 16 + lr;
#pragma unroll
      for (int r = 0; r < 8; ++r)
        Ss[(mt * 16 + half * 8 + r) * 65 + n] = c[r];
    }
    __syncthreads();

    // softmax over the 49 valid columns; 4 threads per row, all 256 active.
    // Cross-thread max/sum via shfl_xor within aligned 4-lane groups (wave32).
    {
      const int row = tid >> 2;
      const int q   = tid & 3;
      if (row < NTOK) {
        float mx = -1e30f;
        for (int c2 = q; c2 < NTOK; c2 += 4) mx = fmaxf(mx, Ss[row * 65 + c2]);
        mx = fmaxf(mx, __shfl_xor(mx, 1, 32));
        mx = fmaxf(mx, __shfl_xor(mx, 2, 32));
        float sum = 0.0f;
        for (int c2 = q; c2 < NTOK; c2 += 4) {
          const float e = __expf(Ss[row * 65 + c2] - mx);
          Ss[row * 65 + c2] = e;
          sum += e;
        }
        sum += __shfl_xor(sum, 1, 32);
        sum += __shfl_xor(sum, 2, 32);
        const float inv = 1.0f / sum;
        for (int c2 = q; c2 < 64; c2 += 4)
          Ps[row * 72 + c2] =
              (c2 < NTOK) ? (__bf16)(Ss[row * 65 + c2] * inv) : (__bf16)0.0f;
      } else {
        for (int c2 = q; c2 < 64; c2 += 4) Ps[row * 72 + c2] = (__bf16)0.0f;
      }
    }
    __syncthreads();

    // O_h = P @ V_h ; 8 tiles (4 M x 2 N), one per wave, K = 64 (2 steps)
    {
      const int mt = wid >> 1, nt = wid & 1;
      v8f c = {};
#pragma unroll
      for (int ks = 0; ks < 2; ++ks) {
        v16bf af = load_a_frag(Ps, 72, lane, mt * 16, ks * 32);
        v16bf bf = load_b_frag(VTs + h * 32 * 72, 72, lane, nt * 16, ks * 32);
        c = wmma_bf16(af, bf, c);
      }
      const int n = h * 32 + nt * 16 + lr;
#pragma unroll
      for (int r = 0; r < 8; ++r)
        Os[(mt * 16 + half * 8 + r) * 136 + n] = (__bf16)c[r];
    }
    __syncthreads();
  }

  // ---------- Y = O @ Wproj + bproj, scatter back to NCHW ----------
  {
    v8f acc[2][4] = {};
    const int nb = wid * 32;
#pragma unroll
    for (int ks = 0; ks < 4; ++ks) {
      v16bf bf0 = load_b_frag(WprojT, 128, lane, nb,      ks * 32);
      v16bf bf1 = load_b_frag(WprojT, 128, lane, nb + 16, ks * 32);
#pragma unroll
      for (int mt = 0; mt < 4; ++mt) {
        v16bf af = load_a_frag(Os, 136, lane, mt * 16, ks * 32);
        acc[0][mt] = wmma_bf16(af, bf0, acc[0][mt]);
        acc[1][mt] = wmma_bf16(af, bf1, acc[1][mt]);
      }
    }
    float* ob = out + (size_t)b * DIM * HWHW;
#pragma unroll
    for (int nt = 0; nt < 2; ++nt) {
      const int n = nb + nt * 16 + lr;
      const float bias = bproj[n];
#pragma unroll
      for (int mt = 0; mt < 4; ++mt) {
#pragma unroll
        for (int r = 0; r < 8; ++r) {
          const int m = mt * 16 + half * 8 + r;
          if (m < NTOK) {
            const int i = (m * 9363) >> 16;
            const int j = m - i * 7;
            ob[(size_t)n * HWHW + (h0 + i) * HW + (w0 + j)] = acc[nt][mt][r] + bias;
          }
        }
      }
    }
  }
}

extern "C" void kernel_launch(void* const* d_in, const int* in_sizes, int n_in,
                              void* d_out, int out_size, void* d_ws, size_t ws_size,
                              hipStream_t stream) {
  (void)in_sizes; (void)n_in; (void)out_size; (void)ws_size;
  const float* x     = (const float*)d_in[0];
  const float* Wqkv  = (const float*)d_in[1];
  const float* bqkv  = (const float*)d_in[2];
  const float* Wproj = (const float*)d_in[3];
  const float* bproj = (const float*)d_in[4];
  float* out = (float*)d_out;

  __bf16* WqkvT  = (__bf16*)d_ws;            // 256*256 bf16 = 128 KB
  __bf16* WprojT = WqkvT + 256 * 256;        // 256*128 bf16 =  64 KB

  prep_weights<<<384, 256, 0, stream>>>(Wqkv, Wproj, WqkvT, WprojT);
  winattn_kernel<<<8192, 256, 0, stream>>>(x, bqkv, bproj, WqkvT, WprojT, out);
}
